// Mamba_43069932044291
// MI455X (gfx1250) — matrix operand order repaired and checked by reading, not compiled, not run on hardware
//
#include <hip/hip_runtime.h>

typedef __attribute__((ext_vector_type(16))) _Float16 v16h;
typedef __attribute__((ext_vector_type(8)))  _Float16 v8h;
typedef __attribute__((ext_vector_type(8)))  float    v8f;

#define BB   8
#define LL   1024
#define DD   768
#define DD2  1536
#define BL   8192   // BB*LL

__device__ __forceinline__ float silu_f(float x)     { return x / (1.f + __expf(-x)); }
__device__ __forceinline__ float softplus_f(float x) { return (x > 20.f) ? x : log1pf(__expf(x)); }

// Async copy of one 16-byte chunk from global to LDS (CDNA5 async path, ASYNCcnt).
__device__ __forceinline__ void async_copy_b128(_Float16* lds_dst, const _Float16* gsrc)
{
    unsigned int       ldsa = (unsigned int)(uintptr_t)lds_dst;     // addrspace(3) offset
    unsigned long long ga   = (unsigned long long)(uintptr_t)gsrc;
    asm volatile("global_load_async_to_lds_b128 %0, %1, off"
                 :: "v"(ldsa), "v"(ga) : "memory");
}
__device__ __forceinline__ void wait_async0()
{
    asm volatile("s_wait_asynccnt 0x0" ::: "memory");
}

// ---------------------------------------------------------------------------
// Embedding gather + per-row partial sums (for per-sample LayerNorm stats)
// ---------------------------------------------------------------------------
__global__ void k_gather(const int* __restrict__ ids, const float* __restrict__ emb,
                         float* __restrict__ xraw, float* __restrict__ partial)
{
    __shared__ float sh[256];
    __shared__ float sh2[256];
    const int row = blockIdx.x;
    const int id  = ids[row];
    const float* src = emb + (size_t)id * DD;
    float s = 0.f, s2 = 0.f;
    for (int d = threadIdx.x; d < DD; d += 256) {
        float v = src[d];
        xraw[(size_t)row * DD + d] = v;
        s += v; s2 += v * v;
    }
    sh[threadIdx.x] = s; sh2[threadIdx.x] = s2;
    __syncthreads();
    for (int st = 128; st > 0; st >>= 1) {
        if (threadIdx.x < st) {
            sh[threadIdx.x]  += sh[threadIdx.x + st];
            sh2[threadIdx.x] += sh2[threadIdx.x + st];
        }
        __syncthreads();
    }
    if (threadIdx.x == 0) {
        partial[row * 2]     = sh[0];
        partial[row * 2 + 1] = sh2[0];
    }
}

// Deterministic per-sample reduction of row partials -> (sum, sumsq) per sample
__global__ void k_reduce_stats(const float* __restrict__ partial, float* __restrict__ stats)
{
    __shared__ float sh[256];
    __shared__ float sh2[256];
    const int b = blockIdx.x;
    float s = 0.f, s2 = 0.f;
    for (int r = threadIdx.x; r < LL; r += 256) {
        s  += partial[(b * LL + r) * 2];
        s2 += partial[(b * LL + r) * 2 + 1];
    }
    sh[threadIdx.x] = s; sh2[threadIdx.x] = s2;
    __syncthreads();
    for (int st = 128; st > 0; st >>= 1) {
        if (threadIdx.x < st) {
            sh[threadIdx.x]  += sh[threadIdx.x + st];
            sh2[threadIdx.x] += sh2[threadIdx.x + st];
        }
        __syncthreads();
    }
    if (threadIdx.x == 0) {
        stats[2 * b]     = sh[0];
        stats[2 * b + 1] = sh2[0];
    }
}

// Per-sample LayerNorm + per-row RMSNorm*rms_w -> xn (f16)
__global__ void k_norm(const float* __restrict__ xraw, const float* __restrict__ stats,
                       const float* __restrict__ rms_w, _Float16* __restrict__ xn)
{
    __shared__ float sh[256];
    const int row = blockIdx.x;
    const int b = row >> 10;
    const float inv = 1.f / (float)(LL * DD);
    const float mu   = stats[2 * b] * inv;
    const float var  = stats[2 * b + 1] * inv - mu * mu;
    const float rsig = rsqrtf(var + 1e-5f);
    float v[3]; float q = 0.f;
    #pragma unroll
    for (int i = 0; i < 3; ++i) {
        int d = threadIdx.x + i * 256;
        float x = xraw[(size_t)row * DD + d];
        float c = (x - mu) * rsig;
        v[i] = c; q += c * c;
    }
    sh[threadIdx.x] = q; __syncthreads();
    for (int st = 128; st > 0; st >>= 1) {
        if (threadIdx.x < st) sh[threadIdx.x] += sh[threadIdx.x + st];
        __syncthreads();
    }
    const float rms = rsqrtf(sh[0] / (float)DD + 1e-5f);
    #pragma unroll
    for (int i = 0; i < 3; ++i) {
        int d = threadIdx.x + i * 256;
        xn[(size_t)row * DD + d] = (_Float16)(v[i] * rms * rms_w[d]);
    }
}

// ---------------------------------------------------------------------------
// Pack f32 weight [K x N] row-major -> f16 WMMA B-fragment order:
//   [(ktile*ntiles + ntile)*32 + lane]*16 + j ,  K(j) = ktile*32 + (lane>>4)*16 + j
// ---------------------------------------------------------------------------
__global__ void k_pack_b(const float* __restrict__ W, _Float16* __restrict__ out, int K, int N)
{
    int t = blockIdx.x * 256 + threadIdx.x;
    int total = (K >> 5) * (N >> 4) * 32;
    if (t >= total) return;
    int lane  = t & 31;
    int tile  = t >> 5;
    int ntiles = N >> 4;
    int ntile = tile % ntiles;
    int ktile = tile / ntiles;
    int hi = lane >> 4, lm = lane & 15;
    int n = ntile * 16 + lm;
    _Float16* dst = out + (size_t)t * 16;
    #pragma unroll
    for (int j = 0; j < 16; ++j) {
        int k = ktile * 32 + hi * 16 + j;
        dst[j] = (_Float16)W[(size_t)k * N + n];
    }
}

// conv_w [1024,1024,3] f32 -> 3 row-major f16 A-matrices cwk[tap][o*1024+i]
__global__ void k_cvt_convw(const float* __restrict__ cw, _Float16* __restrict__ out)
{
    int t = blockIdx.x * 256 + threadIdx.x;
    if (t >= 3 * 1024 * 1024) return;
    int tap  = t >> 20;
    int rest = t & 1048575;
    out[t] = (_Float16)cw[(size_t)rest * 3 + tap];
}

// Repack xp (f16, [b][l=1024 rows][h=1536 cols]) into per-batch B-fragment order
__global__ void k_pack_xp(const _Float16* __restrict__ xp, _Float16* __restrict__ xpB)
{
    int t = blockIdx.x * 256 + threadIdx.x;   // 8 * 32 * 96 * 32 = 786432
    if (t >= 786432) return;
    int lane  = t & 31;
    int rest  = t >> 5;
    int ntile = rest % 96;
    int rest2 = rest / 96;
    int ktile = rest2 & 31;
    int b     = rest2 >> 5;
    int hi = lane >> 4, lm = lane & 15;
    int n = ntile * 16 + lm;
    const _Float16* src = xp + (size_t)b * LL * DD2;
    _Float16* dst = xpB + (size_t)t * 16;
    #pragma unroll
    for (int j = 0; j < 16; ++j) {
        int k = ktile * 32 + hi * 16 + j;
        dst[j] = src[(size_t)k * DD2 + n];
    }
}

// ---------------------------------------------------------------------------
// WMMA f16 GEMM, async-LDS staged B + register-pipelined A:
//   C[MxN] = act(A[MxK] @ Bpacked + bias)
// block = 256 threads (8 waves). Each wave owns a 32 x (NT*16) output tile;
// block covers 256 x (NT*16). The B panel (shared by all 8 waves) is copied
// once per block per k-step into double-buffered LDS with
// global_load_async_to_lds_b128 (ASYNCcnt); A fragments are double-buffered
// in registers so WMMAs of step k overlap the loads of step k+1.
// ---------------------------------------------------------------------------
template<int NT>
__global__ __launch_bounds__(256) void k_gemm(
        const _Float16* __restrict__ A, const _Float16* __restrict__ Bp,
        const float* __restrict__ bias, void* __restrict__ Cout,
        int M, int N, int K, int act, int out_f16)
{
    __shared__ _Float16 lsb[2][NT * 512];

    const int lane = threadIdx.x & 31;
    const int wave = threadIdx.x >> 5;
    const int hi = lane >> 4, lm = lane & 15;
    const int m0 = blockIdx.y * 256 + wave * 32;
    const int n0 = blockIdx.x * (NT * 16);
    const int ntiles = N >> 4;

    v8f acc[2][NT] = {};
    const _Float16* A0 = A + (size_t)(m0 + lm) * K;
    const _Float16* A1 = A + (size_t)(m0 + 16 + lm) * K;

    // issue async copy of panel `kt` into LDS buffer `buf`
    auto copy_panel = [&](int kt, int buf) {
        const int chunks = NT * 64;                       // 8-half (16B) chunks
        if (threadIdx.x < chunks) {
            const _Float16* src = Bp + ((size_t)kt * ntiles + (n0 >> 4)) * 512
                                     + (size_t)threadIdx.x * 8;
            async_copy_b128(&lsb[buf][threadIdx.x * 8], src);
        }
    };

    union AU { v16h v; v8h h[2]; };
    AU a0[2], a1[2];
    auto load_a = [&](int kt, int slot) {
        const int k0 = kt * 32;
        a0[slot].h[0] = *(const v8h*)(A0 + k0 + 16 * hi);
        a0[slot].h[1] = *(const v8h*)(A0 + k0 + 16 * hi + 8);
        a1[slot].h[0] = *(const v8h*)(A1 + k0 + 16 * hi);
        a1[slot].h[1] = *(const v8h*)(A1 + k0 + 16 * hi + 8);
    };

    const int nk = K >> 5;
    copy_panel(0, 0);
    load_a(0, 0);

    for (int kt = 0; kt < nk; ++kt) {
        const int cur = kt & 1;
        wait_async0();          // my async stores into panel[cur] are complete
        __syncthreads();        // everyone's stores done; prev panel fully consumed
        if (kt + 1 < nk) {
            copy_panel(kt + 1, cur ^ 1);    // next B panel -> other LDS buffer
            load_a(kt + 1, cur ^ 1);        // next A frags -> other register slot
        }

        #pragma unroll
        for (int t = 0; t < NT; ++t) {
            v16h bf = *(const v16h*)&lsb[cur][t * 512 + lane * 16];
            acc[0][t] = __builtin_amdgcn_wmma_f32_16x16x32_f16(
                false, a0[cur].v, false, bf, (short)0, acc[0][t], false, false);
            acc[1][t] = __builtin_amdgcn_wmma_f32_16x16x32_f16(
                false, a1[cur].v, false, bf, (short)0, acc[1][t], false, false);
        }
    }

    #pragma unroll
    for (int h = 0; h < 2; ++h) {
        const int mb = m0 + h * 16 + 8 * hi;
        #pragma unroll
        for (int t = 0; t < NT; ++t) {
            int n = n0 + t * 16 + lm;
            float bv = bias ? bias[n] : 0.f;
            #pragma unroll
            for (int r = 0; r < 8; ++r) {
                float v = acc[h][t][r] + bv;
                if (act == 1)      v = silu_f(v);
                else if (act == 2) v = softplus_f(v);
                size_t o = (size_t)(mb + r) * N + n;
                if (out_f16) ((_Float16*)Cout)[o] = (_Float16)v;
                else         ((float*)Cout)[o]    = v;
            }
        }
    }
}

// ---------------------------------------------------------------------------
// Conv1d(L,L,3,pad=1) as 3 accumulated WMMA GEMMs per batch.
//   xc[b,o,h] = sum_tap sum_i cwk[tap][o,i] * xp[b,i,h+tap-1]; then +conv_b[o], silu
// Each wave owns a 32(o) x 64(h) tile; shifted B columns resolved per lane in
// the packed layout (boundary lanes use a zero fragment).
// ---------------------------------------------------------------------------
__global__ __launch_bounds__(256) void k_conv(
        const _Float16* __restrict__ cwk, const _Float16* __restrict__ xpB,
        const float* __restrict__ conv_b, _Float16* __restrict__ xca)
{
    const int lane = threadIdx.x & 31;
    const int wave = threadIdx.x >> 5;
    const int hi = lane >> 4, lm = lane & 15;
    const int m0 = blockIdx.y * 256 + wave * 32;   // o
    const int n0 = blockIdx.x * 64;                // h
    const int b  = blockIdx.z;
    const _Float16* Bb = xpB + (size_t)b * (32 * 96 * 512);

    v8f acc[2][4] = {};
    for (int tap = 0; tap < 3; ++tap) {
        const _Float16* A0 = cwk + (size_t)tap * (1024 * 1024) + (size_t)(m0 + lm) * 1024;
        const _Float16* A1 = A0 + 16 * 1024;
        const int shift = tap - 1;
        for (int k0 = 0; k0 < 1024; k0 += 32) {
            union { v16h v; v8h h[2]; } a0, a1;
            a0.h[0] = *(const v8h*)(A0 + k0 + 16 * hi);
            a0.h[1] = *(const v8h*)(A0 + k0 + 16 * hi + 8);
            a1.h[0] = *(const v8h*)(A1 + k0 + 16 * hi);
            a1.h[1] = *(const v8h*)(A1 + k0 + 16 * hi + 8);
            #pragma unroll
            for (int t = 0; t < 4; ++t) {
                int n = n0 + t * 16 + lm + shift;       // shifted column (h axis)
                v16h bf = {};
                if ((unsigned)n < (unsigned)DD2) {
                    bf = *(const v16h*)(Bb +
                        (((size_t)(k0 >> 5) * 96 + (n >> 4)) * 32 + hi * 16 + (n & 15)) * 16);
                }
                acc[0][t] = __builtin_amdgcn_wmma_f32_16x16x32_f16(
                    false, a0.v, false, bf, (short)0, acc[0][t], false, false);
                acc[1][t] = __builtin_amdgcn_wmma_f32_16x16x32_f16(
                    false, a1.v, false, bf, (short)0, acc[1][t], false, false);
            }
        }
    }

    _Float16* outb = xca + (size_t)b * LL * DD2;
    #pragma unroll
    for (int h = 0; h < 2; ++h) {
        const int mb = m0 + h * 16 + 8 * hi;
        #pragma unroll
        for (int t = 0; t < 4; ++t) {
            int n = n0 + t * 16 + lm;
            #pragma unroll
            for (int r = 0; r < 8; ++r) {
                float v = acc[h][t][r] + conv_b[mb + r];
                outb[(size_t)(mb + r) * DD2 + n] = (_Float16)silu_f(v);
            }
        }
    }
}

// s[row] = sum_n Bm[row,n] * Cm[row,n]
__global__ void k_s(const float* __restrict__ Bm, const float* __restrict__ Cm,
                    float* __restrict__ s)
{
    int r = blockIdx.x * 256 + threadIdx.x;
    if (r < BL) {
        float acc = 0.f;
        #pragma unroll
        for (int n = 0; n < 16; ++n) acc += Bm[r * 16 + n] * Cm[r * 16 + n];
        s[r] = acc;
    }
}

// z = silu(xco * delta * s_row) * x_res   (in-place over the delta buffer)
__global__ void k_fuse(const _Float16* __restrict__ xco, _Float16* __restrict__ dz,
                       const float* __restrict__ s, const _Float16* __restrict__ xres)
{
    size_t idx = (size_t)blockIdx.x * 256 + threadIdx.x;
    if (idx < (size_t)BL * DD2) {
        int row = (int)(idx / DD2);
        float y = (float)xco[idx] * (float)dz[idx] * s[row];
        dz[idx] = (_Float16)(silu_f(y) * (float)xres[idx]);
    }
}

// pooled[b,d] = max_l out[b,l,d]
__global__ void k_pool(const float* __restrict__ out, float* __restrict__ pooled)
{
    int t = blockIdx.x * 256 + threadIdx.x;
    if (t < BB * DD) {
        int b = t / DD, d = t % DD;
        const float* p = out + (size_t)b * LL * DD + d;
        float m = p[0];
        for (int l = 1; l < LL; ++l) m = fmaxf(m, p[(size_t)l * DD]);
        pooled[t] = m;
    }
}

// ---------------------------------------------------------------------------
extern "C" void kernel_launch(void* const* d_in, const int* in_sizes, int n_in,
                              void* d_out, int out_size, void* d_ws, size_t ws_size,
                              hipStream_t stream)
{
    const int*   ids    = (const int*)  d_in[0];
    const float* emb    = (const float*)d_in[1];
    const float* rms_w  = (const float*)d_in[2];
    const float* W_in   = (const float*)d_in[3];
    const float* b_in   = (const float*)d_in[4];
    const float* conv_w = (const float*)d_in[5];
    const float* conv_b = (const float*)d_in[6];
    const float* W_cl   = (const float*)d_in[7];
    const float* b_cl   = (const float*)d_in[8];
    const float* fc1_w  = (const float*)d_in[9];
    const float* fc1_b  = (const float*)d_in[10];
    const float* fc2_w  = (const float*)d_in[11];
    const float* fc2_b  = (const float*)d_in[12];
    const float* fc3_w  = (const float*)d_in[13];
    const float* fc3_b  = (const float*)d_in[14];
    // d_in[15] = A : mathematically unused (zero-initialized SSM state)
    const float* W_D    = (const float*)d_in[16];
    const float* b_D    = (const float*)d_in[17];
    const float* W_out  = (const float*)d_in[18];
    const float* b_out  = (const float*)d_in[19];
    (void)in_sizes; (void)n_in; (void)out_size; (void)ws_size;

    char* ws = (char*)d_ws;
    size_t off = 0;
    auto take = [&](size_t bytes) -> char* {
        char* p = ws + off;
        off = (off + bytes + 255) & ~(size_t)255;
        return p;
    };

    // R0: xraw(f32) -> xp(f16) -> xca(f16) -> delta/z(f16)   (all 25,165,824 B)
    float*    xraw = (float*)take((size_t)BL * DD * 4);
    _Float16* xp   = (_Float16*)xraw;
    _Float16* xca  = (_Float16*)xraw;
    _Float16* dz   = (_Float16*)xraw;
    // R1: xn (f16)
    _Float16* xn   = (_Float16*)take((size_t)BL * DD * 2);
    // R2: xpB (packed) -> xco
    _Float16* xpB  = (_Float16*)take((size_t)BL * DD2 * 2);
    _Float16* xco  = xpB;
    // R3: x_res
    _Float16* xres = (_Float16*)take((size_t)BL * DD2 * 2);
    float*    part  = (float*)take((size_t)BL * 2 * 4);
    float*    stats = (float*)take(64);
    float*    BmB   = (float*)take((size_t)BL * 16 * 4);
    float*    CmB   = (float*)take((size_t)BL * 16 * 4);
    float*    sB    = (float*)take((size_t)BL * 4);
    _Float16* wp_in = (_Float16*)take((size_t)DD  * DD2 * 2);
    _Float16* wp_cl = (_Float16*)take((size_t)DD2 * DD2 * 2);
    _Float16* wp_f1 = (_Float16*)take((size_t)DD2 * DD2 * 2);
    _Float16* wp_f2 = (_Float16*)take((size_t)DD2 * 16 * 2);
    _Float16* wp_f3 = (_Float16*)take((size_t)DD2 * 16 * 2);
    _Float16* wp_D  = (_Float16*)take((size_t)DD  * DD2 * 2);
    _Float16* wp_o  = (_Float16*)take((size_t)DD2 * DD  * 2);
    _Float16* cwk   = (_Float16*)take((size_t)3 * 1024 * 1024 * 2);

    float* outp   = (float*)d_out;
    float* pooled = outp + (size_t)BL * DD;

    // --- weight conversions (constant per call; recomputed for determinism) ---
    auto packb = [&](const float* W, _Float16* dst, int K, int N) {
        int total = (K >> 5) * (N >> 4) * 32;
        k_pack_b<<<(total + 255) / 256, 256, 0, stream>>>(W, dst, K, N);
    };
    packb(W_in,  wp_in, DD,  DD2);
    packb(W_cl,  wp_cl, DD2, DD2);
    packb(fc1_w, wp_f1, DD2, DD2);
    packb(fc2_w, wp_f2, DD2, 16);
    packb(fc3_w, wp_f3, DD2, 16);
    packb(W_D,   wp_D,  DD,  DD2);
    packb(W_out, wp_o,  DD2, DD);
    k_cvt_convw<<<(3 * 1024 * 1024 + 255) / 256, 256, 0, stream>>>(conv_w, cwk);

    // --- embedding + norms ---
    k_gather<<<BL, 256, 0, stream>>>(ids, emb, xraw, part);
    k_reduce_stats<<<BB, 256, 0, stream>>>(part, stats);
    k_norm<<<BL, 256, 0, stream>>>(xraw, stats, rms_w, xn);

    dim3 blk(256);
    // xp = xn @ W_in + b_in
    k_gemm<4><<<dim3(DD2 / 64, BL / 256), blk, 0, stream>>>(xn, wp_in, b_in, xp, BL, DD2, DD, 0, 1);
    // repack xp per batch into B-fragment order for the conv GEMMs
    k_pack_xp<<<(786432 + 255) / 256, 256, 0, stream>>>(xp, xpB);
    // xca = silu(conv(xp) + conv_b)
    k_conv<<<dim3(DD2 / 64, LL / 256, BB), blk, 0, stream>>>(cwk, xpB, conv_b, xca);
    // xco = xca @ W_cl + b_cl
    k_gemm<4><<<dim3(DD2 / 64, BL / 256), blk, 0, stream>>>(xca, wp_cl, b_cl, xco, BL, DD2, DD2, 0, 1);
    // delta = softplus(xco @ fc1_w + fc1_b)
    k_gemm<4><<<dim3(DD2 / 64, BL / 256), blk, 0, stream>>>(xco, wp_f1, fc1_b, dz, BL, DD2, DD2, 2, 1);
    // Bm, Cm (N = 16)
    k_gemm<1><<<dim3(1, BL / 256), blk, 0, stream>>>(xco, wp_f2, fc2_b, BmB, BL, 16, DD2, 0, 0);
    k_gemm<1><<<dim3(1, BL / 256), blk, 0, stream>>>(xco, wp_f3, fc3_b, CmB, BL, 16, DD2, 0, 0);
    k_s<<<(BL + 255) / 256, 256, 0, stream>>>(BmB, CmB, sB);
    // x_res = silu(xn @ W_D + b_D)
    k_gemm<4><<<dim3(DD2 / 64, BL / 256), blk, 0, stream>>>(xn, wp_D, b_D, xres, BL, DD2, DD, 1, 1);
    // z = silu(xco * delta * s) * x_res  (in-place into dz)
    k_fuse<<<(int)(((size_t)BL * DD2 + 255) / 256), 256, 0, stream>>>(xco, dz, sB, xres);
    // out = z @ W_out + b_out  (f32 -> d_out)
    k_gemm<4><<<dim3(DD / 64, BL / 256), blk, 0, stream>>>(dz, wp_o, b_out, outp, BL, DD, DD2, 0, 0);
    // pooled = max over seq
    k_pool<<<(BB * DD + 255) / 256, 256, 0, stream>>>(outp, pooled);
}
